// SLGT2GlobalAttention_71889162600608
// MI455X (gfx1250) — matrix operand order repaired and checked by reading, not compiled, not run on hardware
//
#include <hip/hip_runtime.h>
#include <hip/hip_bf16.h>
#include <math.h>

typedef __attribute__((ext_vector_type(16))) _Float16 v16h;
typedef __attribute__((ext_vector_type(8)))  float    v8f;

// ---------------------------------------------------------------------------
// Problem constants
// ---------------------------------------------------------------------------
#define Bsz   2
#define Dd    16
#define Hh    32
#define Ww    32
#define Cc    384
#define Ntok  (Dd*Hh*Ww)          // 16384
#define NKV   252                 // 1+8+27+216
#define NKP   256                 // padded keys
#define KK    126                 // Nk / TOPK_DIV
#define HDIM  32                  // head dim
#define NHEAD 12
#define LN_EPS 1e-5f
#define ATTN_SCALE 0.17677669529663687f   // 32^-0.5

// ---------------------------------------------------------------------------
// WMMA helpers (CDNA5 v_wmma_f32_16x16x32_f16)
// ---------------------------------------------------------------------------
__device__ __forceinline__ v8f wmma16(v16h a, v16h b, v8f c) {
  return __builtin_amdgcn_wmma_f32_16x16x32_f16(false, a, false, b, (short)0, c, false, false);
}

// A fragment: 16(M) x 32(K) f16, row-major A, stride lda halfs.
// lane<16: M=lane, kb=0 ; lane>=16: M=lane-16, kb=8
// halfs 0..7  = A[M][k0+kb .. k0+kb+7]
// halfs 8..15 = A[M][k0+16+kb .. k0+16+kb+7]
__device__ __forceinline__ v16h load_afrag(const _Float16* A, int lda, int m0, int k0, int lane) {
  int m  = m0 + (lane & 15);
  int kb = (lane < 16) ? 0 : 8;
  const _Float16* p = A + (size_t)m * lda + k0 + kb;
  v16h f;
  ((uint4*)&f)[0] = *(const uint4*)p;
  ((uint4*)&f)[1] = *(const uint4*)(p + 16);
  return f;
}

// B fragment: 32(K) x 16(N) f16, supplied as BT (N x K) row-major, stride ldb halfs.
// lane: N = n0 + lane%16, kb = (lane<16 ? 0 : 8); halfs 0..7 = BT[N][k0+kb .. +7], halfs 8..15 = BT[N][k0+16+kb .. +7]
__device__ __forceinline__ v16h load_bfrag(const _Float16* BT, int ldb, int n0, int k0, int lane) {
  int n  = n0 + (lane & 15);
  int kb = (lane < 16) ? 0 : 8;
  const _Float16* p = BT + (size_t)n * ldb + k0 + kb;
  v16h f;
  ((uint4*)&f)[0] = *(const uint4*)p;
  ((uint4*)&f)[1] = *(const uint4*)(p + 16);
  return f;
}

// ---------------------------------------------------------------------------
// Double-buffered 32(M) x 64(N) x K GEMM tile per wave.
// acc[mi*4+j] = C tile rows [m0+mi*16, +16), cols [tN+j*16, +16)
// ---------------------------------------------------------------------------
__device__ __forceinline__ void gemm_tile_32x64(const _Float16* __restrict__ A, int lda,
                                                const _Float16* __restrict__ BT, int ldb,
                                                int m0, int tN, int K, int lane,
                                                v8f acc[8]) {
  v16h a0 = load_afrag(A, lda, m0,      0, lane);
  v16h a1 = load_afrag(A, lda, m0 + 16, 0, lane);
  v16h b0 = load_bfrag(BT, ldb, tN,      0, lane);
  v16h b1 = load_bfrag(BT, ldb, tN + 16, 0, lane);
  v16h b2 = load_bfrag(BT, ldb, tN + 32, 0, lane);
  v16h b3 = load_bfrag(BT, ldb, tN + 48, 0, lane);
  for (int k0 = 0; k0 < K; k0 += 32) {
    v16h a0n = a0, a1n = a1, b0n = b0, b1n = b1, b2n = b2, b3n = b3;
    int kn = k0 + 32;
    if (kn < K) {                       // issue next iteration's loads first
      a0n = load_afrag(A, lda, m0,      kn, lane);
      a1n = load_afrag(A, lda, m0 + 16, kn, lane);
      b0n = load_bfrag(BT, ldb, tN,      kn, lane);
      b1n = load_bfrag(BT, ldb, tN + 16, kn, lane);
      b2n = load_bfrag(BT, ldb, tN + 32, kn, lane);
      b3n = load_bfrag(BT, ldb, tN + 48, kn, lane);
    }
    acc[0] = wmma16(a0, b0, acc[0]);
    acc[1] = wmma16(a0, b1, acc[1]);
    acc[2] = wmma16(a0, b2, acc[2]);
    acc[3] = wmma16(a0, b3, acc[3]);
    acc[4] = wmma16(a1, b0, acc[4]);
    acc[5] = wmma16(a1, b1, acc[5]);
    acc[6] = wmma16(a1, b2, acc[6]);
    acc[7] = wmma16(a1, b3, acc[7]);
    a0 = a0n; a1 = a1n; b0 = b0n; b1 = b1n; b2 = b2n; b3 = b3n;
  }
}

// ---------------------------------------------------------------------------
// Elementwise utility kernels
// ---------------------------------------------------------------------------
__global__ void k_f32_to_f16(const float* __restrict__ s, _Float16* __restrict__ d, int n) {
  for (int i = blockIdx.x * blockDim.x + threadIdx.x; i < n; i += gridDim.x * blockDim.x)
    d[i] = (_Float16)s[i];
}

__global__ void k_zero_f16(_Float16* __restrict__ d, int n) {
  for (int i = blockIdx.x * blockDim.x + threadIdx.x; i < n; i += gridDim.x * blockDim.x)
    d[i] = (_Float16)0.0f;
}

// ---------------------------------------------------------------------------
// Multi-scale average pooling:  pool[b][p(252)][c]   (c fastest -> coalesced)
// ---------------------------------------------------------------------------
__global__ void k_pool(const float* __restrict__ x, float* __restrict__ pool) {
  int tid = blockIdx.x * blockDim.x + threadIdx.x;
  if (tid >= Bsz * NKV * Cc) return;
  int c = tid % Cc; int rest = tid / Cc;
  int p = rest % NKV; int b = rest / NKV;
  int ri, loc;
  if (p < 1)       { ri = 0; loc = p;      }
  else if (p < 9)  { ri = 1; loc = p - 1;  }
  else if (p < 36) { ri = 2; loc = p - 9;  }
  else             { ri = 3; loc = p - 36; }
  const int R[4] = {1, 2, 3, 6};
  int r = R[ri];
  int e = loc / (r * r), f = (loc / r) % r, g = loc % r;
  int sd = e * Dd / r, ed = ((e + 1) * Dd + r - 1) / r;
  int sh = f * Hh / r, eh = ((f + 1) * Hh + r - 1) / r;
  int sw = g * Ww / r, ew = ((g + 1) * Ww + r - 1) / r;
  float sum = 0.f;
  for (int d = sd; d < ed; ++d)
    for (int h = sh; h < eh; ++h)
      for (int w = sw; w < ew; ++w)
        sum += x[((size_t)b * Ntok + d * (Hh * Ww) + h * Ww + w) * Cc + c];
  float cnt = (float)((ed - sd) * (eh - sh) * (ew - sw));
  pool[((size_t)b * NKV + p) * Cc + c] = sum / cnt;
}

// ---------------------------------------------------------------------------
// Depthwise 3x3x3 'SAME' conv per pooled scale, add residual -> kvx (pre-LN)
// ---------------------------------------------------------------------------
__global__ void k_dwconv(const float* __restrict__ pool,
                         const float* __restrict__ cw,   // (4, C, 27)
                         const float* __restrict__ cb,   // (4, C)
                         float* __restrict__ kvx) {
  int tid = blockIdx.x * blockDim.x + threadIdx.x;
  if (tid >= Bsz * NKV * Cc) return;
  int c = tid % Cc; int rest = tid / Cc;
  int p = rest % NKV; int b = rest / NKV;
  int ri, loc;
  if (p < 1)       { ri = 0; loc = p;      }
  else if (p < 9)  { ri = 1; loc = p - 1;  }
  else if (p < 36) { ri = 2; loc = p - 9;  }
  else             { ri = 3; loc = p - 36; }
  const int R[4]   = {1, 2, 3, 6};
  const int OFF[4] = {0, 1, 9, 36};
  int r = R[ri];
  int e = loc / (r * r), f = (loc / r) % r, g = loc % r;
  float center = pool[((size_t)b * NKV + p) * Cc + c];
  float conv = cb[ri * Cc + c];
  const float* wv = cw + ((size_t)ri * Cc + c) * 27;
  for (int kd = 0; kd < 3; ++kd)
    for (int kh = 0; kh < 3; ++kh)
      for (int kw = 0; kw < 3; ++kw) {
        int ee = e + kd - 1, ff = f + kh - 1, gg = g + kw - 1;
        if (ee >= 0 && ee < r && ff >= 0 && ff < r && gg >= 0 && gg < r) {
          int np = OFF[ri] + (ee * r + ff) * r + gg;
          conv += wv[kd * 9 + kh * 3 + kw] * pool[((size_t)b * NKV + np) * Cc + c];
        }
      }
  kvx[((size_t)b * NKV + p) * Cc + c] = center + conv;
}

// ---------------------------------------------------------------------------
// LayerNorm over C (one wave32 per row), writes f16 A-matrix (512 rows, padded)
// ---------------------------------------------------------------------------
__global__ void __launch_bounds__(32) k_layernorm(const float* __restrict__ kvx,
                                                  const float* __restrict__ gamma,
                                                  const float* __restrict__ beta,
                                                  _Float16* __restrict__ lnh) {
  int m = blockIdx.x;                 // 0..511
  int lane = threadIdx.x;
  if (m >= Bsz * NKV) {               // zero-pad rows 504..511
    for (int j = 0; j < Cc / 32; ++j) lnh[(size_t)m * Cc + lane + j * 32] = (_Float16)0.0f;
    return;
  }
  const float* row = kvx + (size_t)m * Cc;
  float s = 0.f, s2 = 0.f;
  for (int j = 0; j < Cc / 32; ++j) {
    float v = row[lane + j * 32];
    s += v; s2 += v * v;
  }
  for (int off = 16; off > 0; off >>= 1) {
    s  += __shfl_xor(s,  off, 32);
    s2 += __shfl_xor(s2, off, 32);
  }
  float mu  = s / (float)Cc;
  float var = s2 / (float)Cc - mu * mu;
  float inv = rsqrtf(var + LN_EPS);
  for (int j = 0; j < Cc / 32; ++j) {
    int c = lane + j * 32;
    float v = (row[c] - mu) * inv * gamma[c] + beta[c];
    lnh[(size_t)m * Cc + c] = (_Float16)v;
  }
}

// ---------------------------------------------------------------------------
// KV GEMM: (512 x 384) @ (384 x 768) -> scatter K (key,d) and V^T (d,key) f16
// Block = 4 waves, each wave a 32x64 tile -> block tile 128x64.
// ---------------------------------------------------------------------------
__global__ void __launch_bounds__(128) k_gemm_kv(const _Float16* __restrict__ A,
                                                 const _Float16* __restrict__ BT,
                                                 const float* __restrict__ bias,
                                                 _Float16* __restrict__ kf,
                                                 _Float16* __restrict__ vt) {
  int bid = blockIdx.x;
  int tN = (bid % 12) * 64, tM = (bid / 12) * 128;
  int lane = threadIdx.x & 31, wv = threadIdx.x >> 5;
  int m0 = tM + wv * 32;
  v8f acc[8] = {};
  gemm_tile_32x64(A, Cc, BT, Cc, m0, tN, Cc, lane, acc);
  int roff = (lane < 16) ? 0 : 8;
#pragma unroll
  for (int mi = 0; mi < 2; ++mi)
    for (int j = 0; j < 4; ++j)
      for (int r = 0; r < 8; ++r) {
        int m = m0 + mi * 16 + r + roff;
        if (m >= Bsz * NKV) continue;
        int n = tN + j * 16 + (lane & 15);
        float v = acc[mi * 4 + j][r] + bias[n];
        int b = m / NKV, nk = m % NKV;
        int sel = n / Cc, ch = n % Cc;
        int head = ch >> 5, d = ch & 31;
        _Float16 hv = (_Float16)v;
        if (sel == 0) kf[(((size_t)b * NHEAD + head) * NKP + nk) * HDIM + d] = hv;
        else          vt[(((size_t)b * NHEAD + head) * HDIM + d) * NKP + nk] = hv;
      }
}

// ---------------------------------------------------------------------------
// Q GEMM: (32768 x 384) @ (384 x 384) -> scatter f16 into window/head layout
//   g0: [b][w:512][head:6][q:32][d:32]   g1: [b][w:64][head:6][q:256][d:32]
// ---------------------------------------------------------------------------
__global__ void __launch_bounds__(128) k_gemm_q(const _Float16* __restrict__ A,
                                                const _Float16* __restrict__ BT,
                                                _Float16* __restrict__ qh0,
                                                _Float16* __restrict__ qh1) {
  int bid = blockIdx.x;
  int tN = (bid % 6) * 64, tM = (bid / 6) * 128;
  int lane = threadIdx.x & 31, wv = threadIdx.x >> 5;
  int m0 = tM + wv * 32;
  v8f acc[8] = {};
  gemm_tile_32x64(A, Cc, BT, Cc, m0, tN, Cc, lane, acc);
  int roff = (lane < 16) ? 0 : 8;
#pragma unroll
  for (int mi = 0; mi < 2; ++mi)
    for (int j = 0; j < 4; ++j)
      for (int r = 0; r < 8; ++r) {
        int m = m0 + mi * 16 + r + roff;
        int n = tN + j * 16 + (lane & 15);
        int b = m >> 14, tok = m & (Ntok - 1);
        int dp = tok >> 10, hp = (tok >> 5) & 31, wp = tok & 31;
        int gi = n / 192, cg = n % 192;
        int head = cg >> 5, d = cg & 31;
        _Float16 hv = (_Float16)acc[mi * 4 + j][r];
        if (gi == 0) {
          int w = ((dp >> 1) * 8 + (hp >> 2)) * 8 + (wp >> 2);
          int q = ((dp & 1) * 4 + (hp & 3)) * 4 + (wp & 3);
          qh0[((((size_t)b * 512 + w) * 6 + head) * 32 + q) * HDIM + d] = hv;
        } else {
          int w = ((dp >> 2) * 4 + (hp >> 3)) * 4 + (wp >> 3);
          int q = (dp & 3) * 64 + (hp & 7) * 8 + (wp & 7);
          qh1[((((size_t)b * 64 + w) * 6 + head) * 256 + q) * HDIM + d] = hv;
        }
      }
}

// ---------------------------------------------------------------------------
// Attention: one wave32 per (b, window, head, 16-query tile)
//   QK^T via 16 WMMAs -> scores LDS -> exact top-k (binary search on
//   order-preserving u32 map) -> softmax -> attn@V via 16 WMMAs -> scatter f16
// ---------------------------------------------------------------------------
#define SSTR 264   // padded LDS row stride (dwords / halfs) to spread banks

template <int GI>
__global__ void __launch_bounds__(32) k_attn(const _Float16* __restrict__ qh,
                                             const _Float16* __restrict__ kf,
                                             const _Float16* __restrict__ vt,
                                             _Float16* __restrict__ aoh) {
  constexpr int NW = (GI == 0) ? 512 : 64;
  constexpr int NQ = (GI == 0) ? 32  : 256;
  constexpr int QT = (GI == 0) ? 2   : 16;
  constexpr int LD = (GI == 0) ? 2 : 4, LH = (GI == 0) ? 4 : 8, LW = (GI == 0) ? 4 : 8;
  constexpr int GH = (GI == 0) ? 8 : 4, GW = (GI == 0) ? 8 : 4;

  int bi = blockIdx.x;
  int qt   = bi % QT;
  int head = (bi / QT) % 6;
  int w    = (bi / (QT * 6)) % NW;
  int b    = bi / (QT * 6 * NW);
  int lane = threadIdx.x;

  __shared__ __align__(16) float    sc[16 * SSTR];
  __shared__ __align__(16) unsigned mp[16 * SSTR];
  __shared__ __align__(16) _Float16 pr[16 * SSTR];

  int hg = GI * 6 + head;
  const _Float16* qbase = qh + ((((size_t)b * NW + w) * 6 + head) * NQ + qt * 16) * HDIM;
  const _Float16* kbase = kf + (((size_t)b * NHEAD + hg) * NKP) * HDIM;
  const _Float16* vbase = vt + (((size_t)b * NHEAD + hg) * HDIM) * NKP;

  // ---- scores = (Q K^T) * scale ----  (double-buffered key tiles)
  v16h afrag = load_afrag(qbase, HDIM, 0, 0, lane);
  int roff = (lane < 16) ? 0 : 8;
  int coln = lane & 15;
  v16h bcur = load_bfrag(kbase, HDIM, 0, 0, lane);
  for (int nt = 0; nt < NKP / 16; ++nt) {
    v16h bnext = bcur;
    if (nt + 1 < NKP / 16) bnext = load_bfrag(kbase, HDIM, (nt + 1) * 16, 0, lane);
    v8f accs = {};
    accs = wmma16(afrag, bcur, accs);
    int col = nt * 16 + coln;
#pragma unroll
    for (int r = 0; r < 8; ++r) {
      float v = accs[r] * ATTN_SCALE;
      if (col >= NKV) v = -INFINITY;           // kill padded keys
      int row = r + roff;
      sc[row * SSTR + col] = v;
      unsigned u = __float_as_uint(v);
      mp[row * SSTR + col] = (u & 0x80000000u) ? ~u : (u | 0x80000000u);
    }
    bcur = bnext;
  }
  __syncthreads();

  // ---- per-row exact k-th largest (k=126) + softmax over kept entries ----
  if (lane < 16) {
    const unsigned* mrow = &mp[lane * SSTR];
    unsigned lo = 0u, hi = 0xFFFFFFFFu;
    while (lo < hi) {                           // max t: count(>=t) >= KK
      unsigned mid = (unsigned)((((unsigned long long)lo + hi) + 1ull) >> 1);
      int cnt = 0;
      for (int j = 0; j < NKP / 4; ++j) {
        uint4 q4 = *(const uint4*)&mrow[j * 4];
        cnt += (q4.x >= mid) + (q4.y >= mid) + (q4.z >= mid) + (q4.w >= mid);
      }
      if (cnt >= KK) lo = mid; else hi = mid - 1;
    }
    unsigned thrm = lo;
    const float* srow = &sc[lane * SSTR];
    float mx = -INFINITY;
    for (int j = 0; j < NKV; ++j) mx = fmaxf(mx, srow[j]);
    float sum = 0.f;
    for (int j = 0; j < NKP; ++j)
      if (mrow[j] >= thrm) sum += __expf(srow[j] - mx);
    float rs = 1.f / sum;
    for (int j = 0; j < NKP; ++j) {
      float p = (mrow[j] >= thrm) ? __expf(srow[j] - mx) * rs : 0.f;
      pr[lane * SSTR + j] = (_Float16)p;
    }
  }
  __syncthreads();

  // ---- out = attn @ V ----  (double-buffered)
  for (int dt = 0; dt < 2; ++dt) {
    v8f acc = {};
    v16h pa = load_afrag((const _Float16*)pr, SSTR, 0, 0, lane);
    v16h vb = load_bfrag(vbase, NKP, dt * 16, 0, lane);
    for (int kt = 0; kt < NKP / 32; ++kt) {
      v16h pan = pa, vbn = vb;
      if (kt + 1 < NKP / 32) {
        pan = load_afrag((const _Float16*)pr, SSTR, 0, (kt + 1) * 32, lane);
        vbn = load_bfrag(vbase, NKP, dt * 16, (kt + 1) * 32, lane);
      }
      acc = wmma16(pa, vb, acc);
      pa = pan; vb = vbn;
    }
    int d  = dt * 16 + coln;
    int ch = GI * 192 + head * 32 + d;
    int wd = w / (GH * GW), wh = (w / GW) % GH, ww = w % GW;
#pragma unroll
    for (int r = 0; r < 8; ++r) {
      int qg = qt * 16 + r + roff;
      int qd = qg / (LH * LW), qh2 = (qg / LW) % LH, qw2 = qg % LW;
      int dp = wd * LD + qd, hp = wh * LH + qh2, wp = ww * LW + qw2;
      int tok = dp * (Hh * Ww) + hp * Ww + wp;
      aoh[((size_t)b * Ntok + tok) * Cc + ch] = (_Float16)acc[r];
    }
  }
}

// ---------------------------------------------------------------------------
// Output GEMM: (32768 x 384) @ (384 x 384) + bias -> f32 d_out
// ---------------------------------------------------------------------------
__global__ void __launch_bounds__(128) k_gemm_out(const _Float16* __restrict__ A,
                                                  const _Float16* __restrict__ BT,
                                                  const float* __restrict__ bias,
                                                  float* __restrict__ out) {
  int bid = blockIdx.x;
  int tN = (bid % 6) * 64, tM = (bid / 6) * 128;
  int lane = threadIdx.x & 31, wv = threadIdx.x >> 5;
  int m0 = tM + wv * 32;
  v8f acc[8] = {};
  gemm_tile_32x64(A, Cc, BT, Cc, m0, tN, Cc, lane, acc);
  int roff = (lane < 16) ? 0 : 8;
#pragma unroll
  for (int mi = 0; mi < 2; ++mi)
    for (int j = 0; j < 4; ++j)
      for (int r = 0; r < 8; ++r) {
        int m = m0 + mi * 16 + r + roff;
        int n = tN + j * 16 + (lane & 15);
        out[(size_t)m * Cc + n] = acc[mi * 4 + j][r] + bias[n];
      }
}

// ---------------------------------------------------------------------------
// Host-side launch
// ---------------------------------------------------------------------------
extern "C" void kernel_launch(void* const* d_in, const int* in_sizes, int n_in,
                              void* d_out, int out_size, void* d_ws, size_t ws_size,
                              hipStream_t stream) {
  const float* x        = (const float*)d_in[0];
  const float* q_w      = (const float*)d_in[4];
  const float* ppconv_w = (const float*)d_in[5];
  const float* ppconv_b = (const float*)d_in[6];
  const float* pp_gamma = (const float*)d_in[7];
  const float* pp_beta  = (const float*)d_in[8];
  const float* ppproj_w = (const float*)d_in[9];
  const float* ppproj_b = (const float*)d_in[10];
  const float* proj_w   = (const float*)d_in[11];
  const float* proj_b   = (const float*)d_in[12];
  float* out = (float*)d_out;

  char* ws = (char*)d_ws;
  // Workspace layout (bytes; all offsets 256-aligned)
  const size_t NX = (size_t)Bsz * Ntok * Cc;                 // 12,582,912
  _Float16* xh   = (_Float16*)(ws);                          // 25,165,824 B
  _Float16* qwh  = (_Float16*)(ws + 25165824);               //    294,912 B
  _Float16* ppwh = (_Float16*)(ws + 25460736);               //    589,824 B
  _Float16* pwh  = (_Float16*)(ws + 26050560);               //    294,912 B
  float*    pool = (float*)   (ws + 26345472);               //    774,144 B
  float*    kvx  = (float*)   (ws + 27119616);               //    774,144 B
  _Float16* lnh  = (_Float16*)(ws + 27893760);               //    393,216 B
  _Float16* kf   = (_Float16*)(ws + 28286976);               //    393,216 B
  _Float16* vt   = (_Float16*)(ws + 28680192);               //    393,216 B
  _Float16* qh0  = (_Float16*)(ws + 29073408);               // 12,582,912 B
  _Float16* qh1  = (_Float16*)(ws + 41656320);               // 12,582,912 B
  _Float16* aoh  = (_Float16*)(ws + 54239232);               // 25,165,824 B

  // 1) f32 -> f16 conversions (activations + GEMM weights)
  k_f32_to_f16<<<(int)((NX + 255) / 256), 256, 0, stream>>>(x, xh, (int)NX);
  k_f32_to_f16<<<(Cc * Cc + 255) / 256, 256, 0, stream>>>(q_w, qwh, Cc * Cc);
  k_f32_to_f16<<<(2 * Cc * Cc + 255) / 256, 256, 0, stream>>>(ppproj_w, ppwh, 2 * Cc * Cc);
  k_f32_to_f16<<<(Cc * Cc + 255) / 256, 256, 0, stream>>>(proj_w, pwh, Cc * Cc);

  // 2) zero K/V pads (contiguous region)
  k_zero_f16<<<(2 * 196608 + 255) / 256, 256, 0, stream>>>(kf, 2 * 196608);

  // 3) pooling -> depthwise conv + residual -> LayerNorm (f16 A-matrix)
  int npool = Bsz * NKV * Cc;
  k_pool<<<(npool + 255) / 256, 256, 0, stream>>>(x, pool);
  k_dwconv<<<(npool + 255) / 256, 256, 0, stream>>>(pool, ppconv_w, ppconv_b, kvx);
  k_layernorm<<<512, 32, 0, stream>>>(kvx, pp_gamma, pp_beta, lnh);

  // 4) KV projection GEMM (WMMA) -> K (key,d) / V^T (d,key) f16
  k_gemm_kv<<<(512 / 128) * (768 / 64), 128, 0, stream>>>(lnh, ppwh, ppproj_b, kf, vt);

  // 5) Q projection GEMM (WMMA) -> windowed f16 layout
  k_gemm_q<<<((Bsz * Ntok) / 128) * (Cc / 64), 128, 0, stream>>>(xh, qwh, qh0, qh1);

  // 6) windowed top-k attention (WMMA QK^T and attn@V)
  k_attn<0><<<Bsz * 512 * 6 * 2, 32, 0, stream>>>(qh0, kf, vt, aoh);
  k_attn<1><<<Bsz * 64 * 6 * 16, 32, 0, stream>>>(qh1, kf, vt, aoh);

  // 7) output projection GEMM (WMMA) + bias -> f32 d_out
  k_gemm_out<<<((Bsz * Ntok) / 128) * (Cc / 64), 128, 0, stream>>>(aoh, pwh, proj_b, out);
}
